// EncoderRNN_61005715472775
// MI455X (gfx1250) — matrix operand order repaired and checked by reading, not compiled, not run on hardware
//
#include <hip/hip_runtime.h>

// ---------------------------------------------------------------------------
// Bidirectional GRU encoder for MI455X (gfx1250).
//  Kernel 1: pack embeddings + weights to bf16 (E padded 300 -> 320),
//            zero the inter-workgroup barrier counters.
//  Kernel 2: WMMA GEMM  gi[d][t][3H] = x @ Wih_d^T + bih_d
//            (v_wmma_f32_16x16x32_bf16, 2x2 register blocking per wave).
//  Kernel 3: sequential GRU scan, 8 workgroups per direction.  Each WG keeps
//            its 192-row slice of Whh resident in LDS (staged once with
//            global_load_async_to_lds_b128 / s_wait_asynccnt), does the
//            recurrent matvec with WMMA from LDS, and the 8 WGs of a
//            direction sync per step through an L2 atomic barrier.
//  Kernel 4: MLP head + softmax (20 classes).
// ---------------------------------------------------------------------------

typedef unsigned short u16;
typedef unsigned long long u64;
typedef __attribute__((ext_vector_type(16))) __bf16 v16bf;
typedef __attribute__((ext_vector_type(8)))  __bf16 v8bf;
typedef __attribute__((ext_vector_type(8)))  float  v8f;

#define LSEQ 2048
#define EDIM 300
#define EPAD 320            // next multiple of 32
#define HDIM 512
#define G3   1536           // 3*H
#define DDIM 512
#define CCLS 20

#define NWG_DIR 8           // workgroups per direction in the scan
#define ROWS_WG (G3 / NWG_DIR)   // 192 gate rows per workgroup
#define HSL     (HDIM / NWG_DIR) // 64 hidden units owned per workgroup

// float -> bf16 round-to-nearest-even
__device__ __forceinline__ u16 f2bf(float f) {
  unsigned u = __float_as_uint(f);
  u += 0x7FFFu + ((u >> 16) & 1u);
  return (u16)(u >> 16);
}

__device__ __forceinline__ v16bf cat8(v8bf lo, v8bf hi) {
  return __builtin_shufflevector(lo, hi, 0,1,2,3,4,5,6,7,8,9,10,11,12,13,14,15);
}

// ---------------------------------------------------------------------------
// Kernel 1: gather embeddings + pack all weights to bf16 (zero-padded K).
// ---------------------------------------------------------------------------
#define XE_N   (LSEQ * EPAD)
#define WIH_N  (2 * G3 * EPAD)
#define WHH_N  (2 * G3 * HDIM)

__global__ __launch_bounds__(256) void pack_bf16(
    const int*   __restrict__ seq,
    const float* __restrict__ emb,
    const float* __restrict__ wihf, const float* __restrict__ wihb,
    const float* __restrict__ whhf, const float* __restrict__ whhb,
    u16* __restrict__ xe, u16* __restrict__ wih, u16* __restrict__ whh,
    unsigned* __restrict__ barrier_ctr) {
  if (blockIdx.x == 0 && threadIdx.x == 0) {   // reset scan barriers each call
    barrier_ctr[0] = 0;
    barrier_ctr[1] = 0;
  }
  const int total = XE_N + WIH_N + WHH_N;
  for (int idx = blockIdx.x * blockDim.x + threadIdx.x; idx < total;
       idx += gridDim.x * blockDim.x) {
    if (idx < XE_N) {
      int t = idx / EPAD, e = idx % EPAD;
      float v = (e < EDIM) ? emb[(long)seq[t] * EDIM + e] : 0.0f;
      xe[idx] = f2bf(v);
    } else if (idx < XE_N + WIH_N) {
      int r = idx - XE_N;
      int d = r / (G3 * EPAD);
      int q = r % (G3 * EPAD);
      int j = q / EPAD, e = q % EPAD;
      const float* w = d ? wihb : wihf;
      float v = (e < EDIM) ? w[j * EDIM + e] : 0.0f;
      wih[r] = f2bf(v);
    } else {
      int r = idx - XE_N - WIH_N;
      int d = r / (G3 * HDIM);
      int q = r % (G3 * HDIM);
      whh[r] = f2bf(d ? whhb[q] : whhf[q]);
    }
  }
}

// ---------------------------------------------------------------------------
// Kernel 2: gi = xe @ Wih^T + bih, both directions.
// One wave computes a 32x32 tile (2x2 grid of 16x16 WMMA accumulators),
// so each A/B fragment is reused twice.  K loop over EPAD in steps of 32.
// Tiles: M=2048/32=64, N=1536/32=48, dirs=2 -> 6144 waves, 8 waves/block.
// ---------------------------------------------------------------------------
__global__ __launch_bounds__(256) void gi_gemm(
    const u16* __restrict__ xe, const u16* __restrict__ wih_all,
    const float* __restrict__ bih_f, const float* __restrict__ bih_b,
    float* __restrict__ gi_all) {
  const int wave = threadIdx.x >> 5;
  const int lane = threadIdx.x & 31;
  const int tile = blockIdx.x * 8 + wave;
  const int dir  = tile / (64 * 48);
  const int rem  = tile % (64 * 48);
  const int mt = rem / 48, nt = rem % 48;

  const u16*   wih = wih_all + (long)dir * G3 * EPAD;
  float*       gi  = gi_all  + (long)dir * LSEQ * G3;
  const float* bih = dir ? bih_b : bih_f;

  const int l15 = lane & 15;
  const int hi  = lane >> 4;

  // A fragments: lane holds row, K = k + hi*8 + {0..7} and +16 (2x16B loads)
  const u16* a0 = xe + (mt * 32 + l15) * EPAD + hi * 8;
  const u16* a1 = a0 + 16 * EPAD;
  // B fragments (B = Wih^T): lane holds col n, K = k + hi*8 + {0..7} and +16 (2x16B loads)
  const u16* b0 = wih + (nt * 32 + l15) * EPAD + hi * 8;
  const u16* b1 = b0 + 16 * EPAD;

  v8f c00 = {}, c01 = {}, c10 = {}, c11 = {};
#pragma unroll
  for (int k = 0; k < EPAD; k += 32) {
    v16bf A0 = cat8(*(const v8bf*)(a0 + k), *(const v8bf*)(a0 + k + 16));
    v16bf A1 = cat8(*(const v8bf*)(a1 + k), *(const v8bf*)(a1 + k + 16));
    v16bf B0 = cat8(*(const v8bf*)(b0 + k), *(const v8bf*)(b0 + k + 16));
    v16bf B1 = cat8(*(const v8bf*)(b1 + k), *(const v8bf*)(b1 + k + 16));
    c00 = __builtin_amdgcn_wmma_f32_16x16x32_bf16(false, A0, false, B0, (short)0, c00, false, false);
    c01 = __builtin_amdgcn_wmma_f32_16x16x32_bf16(false, A0, false, B1, (short)0, c01, false, false);
    c10 = __builtin_amdgcn_wmma_f32_16x16x32_bf16(false, A1, false, B0, (short)0, c10, false, false);
    c11 = __builtin_amdgcn_wmma_f32_16x16x32_bf16(false, A1, false, B1, (short)0, c11, false, false);
  }

  const int n0  = nt * 32 + l15;         // j=0 column
  const int n1  = n0 + 16;               // j=1 column
  const int mb0 = mt * 32 + hi * 8;      // i=0 rows (D VGPR r -> M = mb + r)
  const int mb1 = mb0 + 16;              // i=1 rows
  const float bv0 = bih[n0], bv1 = bih[n1];
#pragma unroll
  for (int r = 0; r < 8; ++r) {
    gi[(long)(mb0 + r) * G3 + n0] = c00[r] + bv0;
    gi[(long)(mb0 + r) * G3 + n1] = c01[r] + bv1;
    gi[(long)(mb1 + r) * G3 + n0] = c10[r] + bv0;
    gi[(long)(mb1 + r) * G3 + n1] = c11[r] + bv1;
  }
}

// ---------------------------------------------------------------------------
// Kernel 3: sequential GRU scan.  grid = 16 (dir*8 + w), block = 128 (4 waves).
// WG w of direction d owns hidden units [w*64, w*64+64) and the 192 gate rows
// {g*512 + w*64 + i : g<3, i<64}, whose Whh rows live in LDS (192 KB),
// staged once via global_load_async_to_lds_b128.
// Per step: WMMA matvec from LDS (12 row-tiles over 4 waves), gates on 64
// threads, publish bf16 h slice to L2, atomic arrive + spin barrier, reload h.
// ---------------------------------------------------------------------------
__global__ __launch_bounds__(128) void gru_scan(
    const u16*   __restrict__ whh_all,
    const float* __restrict__ gi_all,
    const float* __restrict__ bhh_f, const float* __restrict__ bhh_b,
    u16*   __restrict__ hbuf,          // [2][512] bf16 shared h
    unsigned* __restrict__ barrier_ctr,// [2] per-direction arrive counters
    float* __restrict__ hout) {        // [2][512] final hidden states
  const int dir = blockIdx.x / NWG_DIR;
  const int w   = blockIdx.x % NWG_DIR;

  const u16*   whhd = whh_all + (long)dir * G3 * HDIM;
  const float* gi   = gi_all  + (long)dir * LSEQ * G3;
  const float* bhh  = dir ? bhh_b : bhh_f;
  u16*         hb   = hbuf + dir * HDIM;
  unsigned*    ctr  = barrier_ctr + dir;

  __shared__ u16   wlds[ROWS_WG * HDIM];  // 192 KB resident weight slice
  __shared__ u16   hbfl[HDIM];            // full h in bf16 (B operand)
  __shared__ float ghl[ROWS_WG];          // matvec result (local rows)
  __shared__ float hl[HSL];               // owned h slice, fp32

  const int tid  = threadIdx.x;
  const int wave = tid >> 5;
  const int lane = tid & 31;
  const int l15  = lane & 15;
  const int hi   = lane >> 4;

  // ---- stage Whh slice into LDS with async global->LDS b128 copies --------
  // local row lr -> global gate row R = (lr/64)*512 + w*64 + (lr%64)
  {
    const unsigned wbase = (unsigned)(u64)(&wlds[0]);
    for (int c = tid; c < ROWS_WG * 64; c += 128) {   // 16B chunks, 1024B/row
      const int lr = c >> 6, j = c & 63;
      const int R  = ((lr >> 6) << 9) + w * HSL + (lr & 63);
      const unsigned ldsa = wbase + (unsigned)(lr * 1024 + j * 16);
      const u64 ga = (u64)(whhd + (long)R * HDIM + j * 8);
      asm volatile("global_load_async_to_lds_b128 %0, %1, off"
                   :: "v"(ldsa), "v"(ga) : "memory");
    }
    asm volatile("s_wait_asynccnt 0x0" ::: "memory");
  }
  if (tid < HSL) hl[tid] = 0.0f;
  for (int i = tid; i < HDIM; i += 128) hbfl[i] = 0;
  __syncthreads();

  for (int s = 0; s < LSEQ; ++s) {
    const int t = dir ? (LSEQ - 1 - s) : s;
    const float* git = gi + (long)t * G3;

    // --- phase 1: gh = Whh_slice @ h  (12 local row-tiles, 3 per wave) -----
#pragma unroll
    for (int lti = 0; lti < 3; ++lti) {
      const int lt = wave * 3 + lti;
      const u16* wr = wlds + (lt * 16 + l15) * HDIM + hi * 8;
      v8f acc = {};
#pragma unroll
      for (int kk = 0; kk < HDIM; kk += 32) {
        v16bf a = cat8(*(const v8bf*)(wr + kk), *(const v8bf*)(wr + kk + 16));
        // same 16-K slice of h in every lane -> all 16 D columns identical
        v16bf b = cat8(*(const v8bf*)(hbfl + kk + hi * 8), *(const v8bf*)(hbfl + kk + hi * 8 + 16));
        acc = __builtin_amdgcn_wmma_f32_16x16x32_bf16(
            false, a, false, b, (short)0, acc, false, false);
      }
      if (l15 == 0) {
        const int mb = lt * 16 + hi * 8;
#pragma unroll
        for (int r = 0; r < 8; ++r) ghl[mb + r] = acc[r];
      }
    }
    __syncthreads();

    // --- phase 2: gates on the 64 owned hidden units ------------------------
    if (tid < HSL) {
      const int i = w * HSL + tid;                 // global hidden index
      const float hr = ghl[tid]           + bhh[i];
      const float hz = ghl[HSL + tid]     + bhh[HDIM + i];
      const float hn = ghl[2 * HSL + tid] + bhh[2 * HDIM + i];
      const float ir  = git[i];
      const float iz  = git[HDIM + i];
      const float in_ = git[2 * HDIM + i];
      const float r = 1.0f / (1.0f + __expf(-(ir + hr)));
      const float z = 1.0f / (1.0f + __expf(-(iz + hz)));
      const float n = tanhf(in_ + r * hn);
      const float h = (1.0f - z) * n + z * hl[tid];
      hl[tid] = h;
      hb[i]   = f2bf(h);                           // publish through L2
    }
    __threadfence();
    __syncthreads();

    // --- phase 3: inter-workgroup barrier (8 WGs of this direction) ---------
    if (tid == 0) {
      __hip_atomic_fetch_add(ctr, 1u, __ATOMIC_RELEASE, __HIP_MEMORY_SCOPE_AGENT);
      const unsigned target = (unsigned)(NWG_DIR * (s + 1));
      while (__hip_atomic_load(ctr, __ATOMIC_ACQUIRE, __HIP_MEMORY_SCOPE_AGENT)
             < target) {
        __builtin_amdgcn_s_sleep(1);
      }
    }
    __syncthreads();
    __threadfence();                               // acquire: drop stale lines

    // --- phase 4: refresh full h (bf16) from the shared buffer --------------
    for (int i = tid; i < HDIM; i += 128) hbfl[i] = hb[i];
    __syncthreads();
  }

  if (tid < HSL) hout[dir * HDIM + w * HSL + tid] = hl[tid];
}

// ---------------------------------------------------------------------------
// Kernel 4: MLP head + softmax. One block, 256 threads.
// ---------------------------------------------------------------------------
__global__ __launch_bounds__(256) void head_kernel(
    const float* __restrict__ hcat,                 // [1024] = hf || hb
    const float* __restrict__ W1, const float* __restrict__ b1,
    const float* __restrict__ W2, const float* __restrict__ b2,
    float* __restrict__ out) {
  __shared__ float h[2 * HDIM];
  __shared__ float hid[DDIM];
  __shared__ float lg[CCLS];
  __shared__ float ex[CCLS];

  const int tid = threadIdx.x;
  for (int i = tid; i < 2 * HDIM; i += 256) h[i] = hcat[i];
  __syncthreads();

  for (int j = tid; j < DDIM; j += 256) {
    const float* w = W1 + (long)j * (2 * HDIM);
    float s = b1[j];
    for (int k = 0; k < 2 * HDIM; ++k) s = fmaf(w[k], h[k], s);
    hid[j] = fmaxf(s, 0.0f);
  }
  __syncthreads();

  if (tid < CCLS) {
    const float* w = W2 + (long)tid * DDIM;
    float s = b2[tid];
    for (int k = 0; k < DDIM; ++k) s = fmaf(w[k], hid[k], s);
    lg[tid] = s;
  }
  __syncthreads();

  if (tid < CCLS) {
    float m = lg[0];
    for (int c = 1; c < CCLS; ++c) m = fmaxf(m, lg[c]);
    ex[tid] = __expf(lg[tid] - m);
  }
  __syncthreads();

  if (tid < CCLS) {
    float s = 0.0f;
    for (int c = 0; c < CCLS; ++c) s += ex[c];
    out[tid] = ex[tid] / s;
  }
}

// ---------------------------------------------------------------------------
// Launch. Workspace layout (bytes):
//   xe   bf16 : 2048*320*2        = 1,310,720
//   wih  bf16 : 2*1536*320*2      = 1,966,080
//   whh  bf16 : 2*1536*512*2      = 3,145,728
//   gi   f32  : 2*2048*1536*4     = 25,165,824
//   hout f32  : 2*512*4           = 4,096
//   hbuf bf16 : 2*512*2           = 2,048
//   ctr  u32  : 2*4 (+pad)        = 64              (~31.6 MB total)
// ---------------------------------------------------------------------------
extern "C" void kernel_launch(void* const* d_in, const int* in_sizes, int n_in,
                              void* d_out, int out_size, void* d_ws, size_t ws_size,
                              hipStream_t stream) {
  (void)in_sizes; (void)n_in; (void)out_size; (void)ws_size;

  const int*   seq   = (const int*)  d_in[0];
  const float* emb   = (const float*)d_in[1];
  const float* Wih_f = (const float*)d_in[2];
  const float* Whh_f = (const float*)d_in[3];
  const float* bih_f = (const float*)d_in[4];
  const float* bhh_f = (const float*)d_in[5];
  const float* Wih_b = (const float*)d_in[6];
  const float* Whh_b = (const float*)d_in[7];
  const float* bih_b = (const float*)d_in[8];
  const float* bhh_b = (const float*)d_in[9];
  const float* W1    = (const float*)d_in[10];
  const float* b1    = (const float*)d_in[11];
  const float* W2    = (const float*)d_in[12];
  const float* b2    = (const float*)d_in[13];

  u16*      xe   = (u16*)d_ws;
  u16*      wih  = xe  + (size_t)LSEQ * EPAD;
  u16*      whh  = wih + (size_t)2 * G3 * EPAD;
  float*    gi   = (float*)(whh + (size_t)2 * G3 * HDIM);
  float*    hout = gi + (size_t)2 * LSEQ * G3;
  u16*      hbuf = (u16*)(hout + 2 * HDIM);
  unsigned* ctr  = (unsigned*)(hbuf + 2 * HDIM);

  pack_bf16<<<2048, 256, 0, stream>>>(seq, emb, Wih_f, Wih_b, Whh_f, Whh_b,
                                      xe, wih, whh, ctr);
  gi_gemm<<<768, 256, 0, stream>>>(xe, wih, bih_f, bih_b, gi);
  gru_scan<<<2 * NWG_DIR, 128, 0, stream>>>(whh, gi, bhh_f, bhh_b,
                                            hbuf, ctr, hout);
  head_kernel<<<1, 256, 0, stream>>>(hout, W1, b1, W2, b2, (float*)d_out);
}
